// MultiHeadAttenInduction_11149735100888
// MI455X (gfx1250) — compile-verified, matrix operand order checked by reading, not run on hardware
//
#include <hip/hip_runtime.h>
#include <hip/hip_bf16.h>

// ---------------- types ----------------
typedef __bf16 bf16_t;
typedef __attribute__((ext_vector_type(16))) __bf16 v16bf;
typedef __attribute__((ext_vector_type(8)))  __bf16 v8bf;
typedef __attribute__((ext_vector_type(8)))  float  v8f;
typedef __attribute__((ext_vector_type(4)))  float  v4f;

#define NEGBIG  (-1.0e8f)
#define B_      8
#define LQ_     512
#define LK_     2048
#define NH_     16
#define D_      64
#define HID_    1024

// CDNA5 16-bit A (16x32) / B-from-column-major fragment:
// lane L: row = L&15 ; khalf = L>>4 ; element i -> k = koff + (i<8?0:16) + 8*khalf + (i&7)
__device__ __forceinline__ v16bf frag_bf16(const bf16_t* base, int ld, int koff) {
  const int lane = threadIdx.x & 31;
  const int row  = lane & 15;
  const int kh   = (lane >> 4) << 3;
  const bf16_t* p = base + row * ld + koff + kh;
  v8bf lo = *(const v8bf*)(p);
  v8bf hi = *(const v8bf*)(p + 16);
  v16bf f;
#pragma unroll
  for (int i = 0; i < 8; ++i) { f[i] = lo[i]; f[i + 8] = hi[i]; }
  return f;
}

// same fragment, fp32 source with on-the-fly (hardware) bf16 conversion
__device__ __forceinline__ v16bf frag_f32(const float* base, int ld, int koff) {
  const int lane = threadIdx.x & 31;
  const int row  = lane & 15;
  const int kh   = (lane >> 4) << 3;
  const float* p = base + row * ld + koff + kh;
  v4f a0 = *(const v4f*)(p);
  v4f a1 = *(const v4f*)(p + 4);
  v4f b0 = *(const v4f*)(p + 16);
  v4f b1 = *(const v4f*)(p + 20);
  v16bf f;
#pragma unroll
  for (int i = 0; i < 4; ++i) {
    f[i]      = (__bf16)a0[i];
    f[i + 4]  = (__bf16)a1[i];
    f[i + 8]  = (__bf16)b0[i];
    f[i + 12] = (__bf16)b1[i];
  }
  return f;
}

__device__ __forceinline__ v8f wmma_bf16(v16bf a, v16bf b, v8f c) {
  return __builtin_amdgcn_wmma_f32_16x16x32_bf16(false, a, false, b, (short)0, c, false, false);
}

// ------------- kernel 0: fp32 -> bf16 conversion (weights) -------------
__global__ void cvt_bf16(const float* __restrict__ src, bf16_t* __restrict__ dst, int n) {
  int i = blockIdx.x * blockDim.x + threadIdx.x;
  int stride = gridDim.x * blockDim.x;
  for (; i < n; i += stride) dst[i] = (__bf16)src[i];
}

// ------------- kernel 1: Q/K projection (X(R,64) @ W^T -> bf16 row-major) -------------
__global__ void proj16(const float* __restrict__ X, const bf16_t* __restrict__ Wb,
                       bf16_t* __restrict__ Out, int nrows) {
  const int gw   = (blockIdx.x * blockDim.x + threadIdx.x) >> 5;
  const int lane = threadIdx.x & 31;
  if (gw * 16 >= nrows) return;
  const float* xbase = X + (size_t)gw * 16 * D_;
  v16bf a0 = frag_f32(xbase, D_, 0);
  v16bf a1 = frag_f32(xbase, D_, 32);
  const int hf = lane >> 4, n = lane & 15;
#pragma unroll
  for (int nt = 0; nt < 4; ++nt) {
    v8f c = {};
    c = wmma_bf16(a0, frag_bf16(Wb + nt * 16 * D_, D_, 0),  c);
    c = wmma_bf16(a1, frag_bf16(Wb + nt * 16 * D_, D_, 32), c);
#pragma unroll
    for (int r = 0; r < 8; ++r) {
      int m = r + 8 * hf;
      Out[(size_t)(gw * 16 + m) * D_ + nt * 16 + n] = (__bf16)c[r];
    }
  }
}

// ------------- kernel 2: V projection, stored transposed Vt[b][h][d][key] -------------
__global__ void proj16_vT(const float* __restrict__ X, const bf16_t* __restrict__ Wb,
                          bf16_t* __restrict__ Vt) {
  const int gw   = (blockIdx.x * blockDim.x + threadIdx.x) >> 5;   // one (b, key) token
  const int lane = threadIdx.x & 31;
  const int b = gw / LK_, l = gw % LK_;
  const float* xbase = X + (size_t)gw * 16 * D_;   // rows = 16 heads of token (b,l)
  v16bf a0 = frag_f32(xbase, D_, 0);
  v16bf a1 = frag_f32(xbase, D_, 32);
  const int hf = lane >> 4, n = lane & 15;
#pragma unroll
  for (int nt = 0; nt < 4; ++nt) {
    v8f c = {};
    c = wmma_bf16(a0, frag_bf16(Wb + nt * 16 * D_, D_, 0),  c);
    c = wmma_bf16(a1, frag_bf16(Wb + nt * 16 * D_, D_, 32), c);
#pragma unroll
    for (int r = 0; r < 8; ++r) {
      int h = r + 8 * hf;                 // C-layout row = head
      int d = nt * 16 + n;
      Vt[(((size_t)b * NH_ + h) * D_ + d) * LK_ + l] = (__bf16)c[r];
    }
  }
}

// ------------- kernel 3: flash attention, one wave per (b, h, 16 q-rows) -------------
// 64-key chunks: shuffle reductions and O-rescale are per-chunk costs, so the wider
// chunk halves softmax VALU per key while WMMA work per key is unchanged.
__global__ void flash_attn(const bf16_t* __restrict__ Qb,
                           const bf16_t* __restrict__ Kb,
                           const bf16_t* __restrict__ Vt,
                           const float* __restrict__ kg_mask,
                           bf16_t* __restrict__ ctx) {
  __shared__ __attribute__((aligned(16))) bf16_t psh[8][16 * 64];
  const int wave = threadIdx.x >> 5, lane = threadIdx.x & 31;
  const int gw = blockIdx.x * 8 + wave;
  const int qt = gw & 31;                 // 32 q-tiles
  const int h  = (gw >> 5) & 15;
  const int b  = gw >> 9;
  const int hf = lane >> 4, n = lane & 15;

  const bf16_t* qbase = Qb + ((size_t)(b * LQ_ + qt * 16) * NH_ + h) * D_;
  v16bf qf0 = frag_bf16(qbase, NH_ * D_, 0);
  v16bf qf1 = frag_bf16(qbase, NH_ * D_, 32);

  v8f o0 = {}, o1 = {}, o2 = {}, o3 = {};
  float mrow[8], lrow[8];
#pragma unroll
  for (int r = 0; r < 8; ++r) { mrow[r] = -3.0e38f; lrow[r] = 0.0f; }

  const bf16_t* vtb = Vt + (((size_t)b * NH_ + h) * D_) * LK_;

  for (int kc = 0; kc < LK_; kc += 64) {
    const bf16_t* kb = Kb + ((size_t)(b * LK_ + kc) * NH_ + h) * D_;
    const size_t kstride = (size_t)16 * NH_ * D_;
    v8f s0 = {}, s1 = {}, s2 = {}, s3 = {};
    s0 = wmma_bf16(qf0, frag_bf16(kb + 0 * kstride, NH_ * D_, 0),  s0);
    s0 = wmma_bf16(qf1, frag_bf16(kb + 0 * kstride, NH_ * D_, 32), s0);
    s1 = wmma_bf16(qf0, frag_bf16(kb + 1 * kstride, NH_ * D_, 0),  s1);
    s1 = wmma_bf16(qf1, frag_bf16(kb + 1 * kstride, NH_ * D_, 32), s1);
    s2 = wmma_bf16(qf0, frag_bf16(kb + 2 * kstride, NH_ * D_, 0),  s2);
    s2 = wmma_bf16(qf1, frag_bf16(kb + 2 * kstride, NH_ * D_, 32), s2);
    s3 = wmma_bf16(qf0, frag_bf16(kb + 3 * kstride, NH_ * D_, 0),  s3);
    s3 = wmma_bf16(qf1, frag_bf16(kb + 3 * kstride, NH_ * D_, 32), s3);

    const float km0 = (1.0f - kg_mask[b * LK_ + kc + n])      * NEGBIG;
    const float km1 = (1.0f - kg_mask[b * LK_ + kc + 16 + n]) * NEGBIG;
    const float km2 = (1.0f - kg_mask[b * LK_ + kc + 32 + n]) * NEGBIG;
    const float km3 = (1.0f - kg_mask[b * LK_ + kc + 48 + n]) * NEGBIG;

#pragma unroll
    for (int r = 0; r < 8; ++r) {
      float xa = s0[r] * 0.125f + km0;        // 1/sqrt(64) scale + key-side mask
      float xb = s1[r] * 0.125f + km1;        // (query-side mask cancels in softmax)
      float xc = s2[r] * 0.125f + km2;
      float xd = s3[r] * 0.125f + km3;
      float mx = fmaxf(fmaxf(xa, xb), fmaxf(xc, xd));
      mx = fmaxf(mx, __shfl_xor(mx, 1, 32));
      mx = fmaxf(mx, __shfl_xor(mx, 2, 32));
      mx = fmaxf(mx, __shfl_xor(mx, 4, 32));
      mx = fmaxf(mx, __shfl_xor(mx, 8, 32));
      float mnew = fmaxf(mrow[r], mx);
      float corr = __expf(mrow[r] - mnew);
      mrow[r] = mnew;
      float p0 = __expf(xa - mnew);
      float p1 = __expf(xb - mnew);
      float p2 = __expf(xc - mnew);
      float p3 = __expf(xd - mnew);
      float rs = (p0 + p1) + (p2 + p3);
      rs += __shfl_xor(rs, 1, 32);
      rs += __shfl_xor(rs, 2, 32);
      rs += __shfl_xor(rs, 4, 32);
      rs += __shfl_xor(rs, 8, 32);
      lrow[r] = lrow[r] * corr + rs;
      o0[r] *= corr; o1[r] *= corr; o2[r] *= corr; o3[r] *= corr;
      int m = r + 8 * hf;                     // C-layout -> row-major P tile in LDS
      psh[wave][m * 64 + n]      = (__bf16)p0;
      psh[wave][m * 64 + 16 + n] = (__bf16)p1;
      psh[wave][m * 64 + 32 + n] = (__bf16)p2;
      psh[wave][m * 64 + 48 + n] = (__bf16)p3;
    }
    __builtin_amdgcn_wave_barrier();          // LDS in-order per wave; stop reordering
    v16bf pf0 = frag_bf16(&psh[wave][0], 64, 0);
    v16bf pf1 = frag_bf16(&psh[wave][0], 64, 32);
    __builtin_amdgcn_wave_barrier();
    o0 = wmma_bf16(pf0, frag_bf16(vtb + (size_t)(0 * 16) * LK_ + kc, LK_, 0), o0);
    o1 = wmma_bf16(pf0, frag_bf16(vtb + (size_t)(1 * 16) * LK_ + kc, LK_, 0), o1);
    o2 = wmma_bf16(pf0, frag_bf16(vtb + (size_t)(2 * 16) * LK_ + kc, LK_, 0), o2);
    o3 = wmma_bf16(pf0, frag_bf16(vtb + (size_t)(3 * 16) * LK_ + kc, LK_, 0), o3);
    o0 = wmma_bf16(pf1, frag_bf16(vtb + (size_t)(0 * 16) * LK_ + kc + 32, LK_, 0), o0);
    o1 = wmma_bf16(pf1, frag_bf16(vtb + (size_t)(1 * 16) * LK_ + kc + 32, LK_, 0), o1);
    o2 = wmma_bf16(pf1, frag_bf16(vtb + (size_t)(2 * 16) * LK_ + kc + 32, LK_, 0), o2);
    o3 = wmma_bf16(pf1, frag_bf16(vtb + (size_t)(3 * 16) * LK_ + kc + 32, LK_, 0), o3);
  }

#pragma unroll
  for (int r = 0; r < 8; ++r) {
    float inv = 1.0f / lrow[r];
    int m = r + 8 * hf;
    size_t o = (size_t)(b * LQ_ + qt * 16 + m) * HID_ + h * D_ + n;
    ctx[o + 0]  = (__bf16)(o0[r] * inv);
    ctx[o + 16] = (__bf16)(o1[r] * inv);
    ctx[o + 32] = (__bf16)(o2[r] * inv);
    ctx[o + 48] = (__bf16)(o3[r] * inv);
  }
}

// ------------- kernel 4: lin = ctx @ W_lin^T + b (pure bf16 fragments) -------------
__global__ void lin_gemm(const bf16_t* __restrict__ ctx,
                         const bf16_t* __restrict__ Wb, const float* __restrict__ bias,
                         float* __restrict__ lin) {
  const int gw   = (blockIdx.x * blockDim.x + threadIdx.x) >> 5;
  const int lane = threadIdx.x & 31;
  const int mt   = gw & 255;                 // 256 row tiles of 16
  const int nt64 = gw >> 8;                  // 16 column groups of 64
  const int hf = lane >> 4, n = lane & 15;
  const bf16_t* abase = ctx + (size_t)mt * 16 * HID_;
  const bf16_t* wb = Wb + (size_t)nt64 * 64 * HID_;
  v8f c0 = {}, c1 = {}, c2 = {}, c3 = {};
  for (int ks = 0; ks < HID_; ks += 32) {
    v16bf a = frag_bf16(abase, HID_, ks);
    c0 = wmma_bf16(a, frag_bf16(wb + (size_t)0 * 16 * HID_, HID_, ks), c0);
    c1 = wmma_bf16(a, frag_bf16(wb + (size_t)1 * 16 * HID_, HID_, ks), c1);
    c2 = wmma_bf16(a, frag_bf16(wb + (size_t)2 * 16 * HID_, HID_, ks), c2);
    c3 = wmma_bf16(a, frag_bf16(wb + (size_t)3 * 16 * HID_, HID_, ks), c3);
  }
#pragma unroll
  for (int r = 0; r < 8; ++r) {
    int m = r + 8 * hf;
    size_t o = (size_t)(mt * 16 + m) * HID_ + nt64 * 64 + n;
    lin[o + 0]  = c0[r] + bias[nt64 * 64 + n];
    lin[o + 16] = c1[r] + bias[nt64 * 64 + 16 + n];
    lin[o + 32] = c2[r] + bias[nt64 * 64 + 32 + n];
    lin[o + 48] = c3[r] + bias[nt64 * 64 + 48 + n];
  }
}

// ------------- kernel 5: exact GELU + residual + LayerNorm -------------
__global__ void gelu_ln(const float* __restrict__ lin, const float* __restrict__ x0,
                        const float* __restrict__ gamma, const float* __restrict__ beta,
                        float* __restrict__ out) {
  __shared__ float shs[8], shs2[8];
  const int row = blockIdx.x;
  const int t = threadIdx.x;
  float v[4];
  float s = 0.0f, s2 = 0.0f;
#pragma unroll
  for (int i = 0; i < 4; ++i) {
    int c = t + i * 256;
    float l = lin[(size_t)row * HID_ + c];
    float g = 0.5f * l * (1.0f + erff(l * 0.70710678118654752f));  // exact erf GELU
    float x = g + x0[(size_t)row * HID_ + c];
    v[i] = x; s += x; s2 += x * x;
  }
#pragma unroll
  for (int msk = 1; msk < 32; msk <<= 1) {
    s  += __shfl_xor(s,  msk, 32);
    s2 += __shfl_xor(s2, msk, 32);
  }
  if ((t & 31) == 0) { shs[t >> 5] = s; shs2[t >> 5] = s2; }
  __syncthreads();
  float ts = 0.0f, ts2 = 0.0f;
#pragma unroll
  for (int i = 0; i < 8; ++i) { ts += shs[i]; ts2 += shs2[i]; }
  const float mu   = ts * (1.0f / HID_);
  const float var  = ts2 * (1.0f / HID_) - mu * mu;
  const float rstd = rsqrtf(var + 1e-5f);
#pragma unroll
  for (int i = 0; i < 4; ++i) {
    int c = t + i * 256;
    out[(size_t)row * HID_ + c] = (v[i] - mu) * rstd * gamma[c] + beta[c];
  }
}

// ---------------- host launcher ----------------
extern "C" void kernel_launch(void* const* d_in, const int* in_sizes, int n_in,
                              void* d_out, int out_size, void* d_ws, size_t ws_size,
                              hipStream_t stream) {
  (void)in_sizes; (void)n_in; (void)out_size; (void)ws_size;
  const float* input_embed = (const float*)d_in[0];
  const float* kg_embed    = (const float*)d_in[1];
  // d_in[2] = input_mask: a per-query additive constant shifts all logits of a row
  // equally and cancels exactly in softmax -> not needed.
  const float* kg_mask = (const float*)d_in[3];
  const float* Wq   = (const float*)d_in[4];
  const float* Wk   = (const float*)d_in[5];
  const float* Wv   = (const float*)d_in[6];
  const float* Wlin = (const float*)d_in[7];
  const float* blin = (const float*)d_in[8];
  const float* gam  = (const float*)d_in[9];
  const float* bet  = (const float*)d_in[10];

  char* ws = (char*)d_ws;
  bf16_t* Qb  = (bf16_t*)(ws);                        //  8 MiB
  bf16_t* Kb  = (bf16_t*)(ws + ((size_t)8  << 20));   // 32 MiB
  bf16_t* Vt  = (bf16_t*)(ws + ((size_t)40 << 20));   // 32 MiB
  bf16_t* Cb  = (bf16_t*)(ws + ((size_t)72 << 20));   //  8 MiB
  float*  Ln  = (float*) (ws + ((size_t)80 << 20));   // 16 MiB
  bf16_t* Wlb = (bf16_t*)(ws + ((size_t)96 << 20));   //  2 MiB
  bf16_t* Wqb = (bf16_t*)(ws + ((size_t)98 << 20));
  bf16_t* Wkb = Wqb + D_ * D_;
  bf16_t* Wvb = Wkb + D_ * D_;

  cvt_bf16<<<1024, 256, 0, stream>>>(Wlin, Wlb, HID_ * HID_);
  cvt_bf16<<<   4, 256, 0, stream>>>(Wq,   Wqb, D_ * D_);
  cvt_bf16<<<   4, 256, 0, stream>>>(Wk,   Wkb, D_ * D_);
  cvt_bf16<<<   4, 256, 0, stream>>>(Wv,   Wvb, D_ * D_);

  proj16   <<< 512, 256, 0, stream>>>(input_embed, Wqb, Qb, B_ * LQ_ * NH_);
  proj16   <<<2048, 256, 0, stream>>>(kg_embed,    Wkb, Kb, B_ * LK_ * NH_);
  proj16_vT<<<2048, 256, 0, stream>>>(kg_embed,    Wvb, Vt);
  flash_attn<<<512, 256, 0, stream>>>(Qb, Kb, Vt, kg_mask, Cb);
  lin_gemm <<< 512, 256, 0, stream>>>(Cb, Wlb, blin, Ln);
  gelu_ln  <<<4096, 256, 0, stream>>>(Ln, input_embed, gam, bet, (float*)d_out);
}
